// CausalSelfAttention_5720896438369
// MI455X (gfx1250) — compile-verified, matrix operand order checked by reading, not run on hardware
//
#include <hip/hip_runtime.h>
#include <hip/hip_bf16.h>

// ---------------------------------------------------------------------------
// CausalSelfAttention for MI455X (gfx1250, wave32, WMMA)
// All matmuls via v_wmma_f32_16x16x32_bf16 (bf16 A/B, f32 accum).
// - GEMMs: 64x64 wave tiles -> 1 b128 load per wmma issue.
// - Attention: S^T = K*Q^T so P feeds the P*V WMMA A-fragment from registers;
//   V pre-transposed by the QKV epilogue; 32 q-rows per wave so K/V fragments
//   are shared by two 16-row score tiles (1 load : 1 wmma).
// ---------------------------------------------------------------------------

typedef __attribute__((ext_vector_type(16))) __bf16 v16bf;
typedef __attribute__((ext_vector_type(8)))  float  v8f;

union FragBF {            // 16 bf16 = one WMMA A or B fragment per lane
    v16bf v;
    uint4 u[2];
};
union Pack8 {             // 8 bf16 packed for one b128 store
    unsigned short us[8];
    uint4 u;
};

__device__ __forceinline__ v8f wmma_bf16(const FragBF& a, const FragBF& b, v8f c) {
    return __builtin_amdgcn_wmma_f32_16x16x32_bf16(
        /*neg_a=*/false, a.v, /*neg_b=*/false, b.v,
        /*c_mod=*/(short)0, c, /*reuse_a=*/false, /*reuse_b=*/false);
}

// dims
#define BATCH 2
#define SEQ   2048
#define DMODEL 1024
#define NHEAD 16
#define DK    64
#define MROWS (BATCH*SEQ)          // 4096
#define NQKV  (3*DMODEL)           // 3072

// ---------------------------------------------------------------------------
// fp32 -> bf16 elementwise convert
// ---------------------------------------------------------------------------
__global__ void cvt_f32_bf16(const float* __restrict__ src,
                             __bf16* __restrict__ dst, int n) {
    int i = blockIdx.x * blockDim.x + threadIdx.x;
    if (i < n) dst[i] = (__bf16)src[i];
}

// ---------------------------------------------------------------------------
// QKV GEMM: C[M=4096, N=3072] = Xbf[M,K] * Wbf[N,K]^T + bias
// Epilogue: n in [0,2048)   -> qk[m, n]       (Q rows then K rows, bf16)
//           n in [2048,3072)-> vt[b,h,d, s]   (V transposed, contiguous b128)
// Block: 256 thr = 8 waves in 2x4, block tile 128x256, wave tile 64x64.
// ---------------------------------------------------------------------------
__global__ __launch_bounds__(256) void qkv_gemm(
    const __bf16* __restrict__ X,    // [4096,1024]
    const __bf16* __restrict__ W,    // [3072,1024]
    const float*  __restrict__ bias, // [3072]
    __bf16* __restrict__ qk,         // [4096,2048]
    __bf16* __restrict__ vt)         // [B*H*64, 2048]
{
    const int K = 1024;
    int lane = threadIdx.x & 31;
    int wv   = threadIdx.x >> 5;
    int lr = lane & 15, lh = lane >> 4;
    int wm = wv >> 2, wn = wv & 3;
    int m0 = blockIdx.y * 128 + wm * 64;
    int n0 = blockIdx.x * 256 + wn * 64;

    const __bf16* arow[4];
    const __bf16* brow[4];
#pragma unroll
    for (int i = 0; i < 4; ++i) {
        arow[i] = X + (size_t)(m0 + i*16 + lr) * K + lh*8;
        brow[i] = W + (size_t)(n0 + i*16 + lr) * K + lh*8;
    }

    v8f acc[4][4] = {};

    for (int k0 = 0; k0 < K; k0 += 32) {
        FragBF a[4], bf[4];
#pragma unroll
        for (int i = 0; i < 4; ++i) {
            a[i].u[0]  = *(const uint4*)(arow[i] + k0);
            a[i].u[1]  = *(const uint4*)(arow[i] + k0 + 16);
            bf[i].u[0] = *(const uint4*)(brow[i] + k0);
            bf[i].u[1] = *(const uint4*)(brow[i] + k0 + 16);
        }
#pragma unroll
        for (int mi = 0; mi < 4; ++mi)
#pragma unroll
            for (int ni = 0; ni < 4; ++ni)
                acc[mi][ni] = wmma_bf16(a[mi], bf[ni], acc[mi][ni]);
    }

    // epilogue: C layout = lane holds column n, rows m = tileM + r + 8*lh
#pragma unroll
    for (int mi = 0; mi < 4; ++mi) {
        int mbase = m0 + mi * 16;
#pragma unroll
        for (int ni = 0; ni < 4; ++ni) {
            int n  = n0 + ni*16 + lr;
            float bv = bias[n];
            if (n < 2048) {                       // Q or K -> row-major bf16
#pragma unroll
                for (int r = 0; r < 8; ++r) {
                    int m = mbase + r + 8*lh;
                    qk[(size_t)m * 2048 + n] = (__bf16)(acc[mi][ni][r] + bv);
                }
            } else {                              // V -> transposed, b128
                int hh = (n - 2048) >> 6;
                int d  = (n - 2048) & 63;
                int bb = mbase >> 11;             // m / SEQ
                int sbase = (mbase + 8*lh) & 2047;
                Pack8 pk;
#pragma unroll
                for (int r = 0; r < 8; ++r)
                    pk.us[r] = __builtin_bit_cast(unsigned short,
                                   (__bf16)(acc[mi][ni][r] + bv));
                *(uint4*)(vt + ((size_t)(bb*NHEAD + hh)*DK + d)*SEQ + sbase) = pk.u;
            }
        }
    }
}

// ---------------------------------------------------------------------------
// Flash-style causal attention. One wave per (b, h, 32-row q block).
// Scores computed transposed: S^T = K * Q^T, so P lives in registers in
// exactly the A-fragment layout needed for O = P * V. The two 16-row q
// halves share every K and V fragment load (q0 multiple of 32 => identical
// causal loop bounds for both halves).
// ---------------------------------------------------------------------------
__global__ __launch_bounds__(32) void attn_kernel(
    const __bf16* __restrict__ qk,   // [4096,2048] : Q cols 0..1023, K 1024..2047
    const __bf16* __restrict__ vt,   // [B*H*64, 2048]
    __bf16* __restrict__ attn)       // [4096,1024]
{
    const float SC = 0.125f;              // 1/sqrt(64), folded into exp
    int lane = threadIdx.x & 31;
    int lr = lane & 15, lh = lane >> 4;

    int idx = blockIdx.x;
    int q0 = (idx & 63) * 32;
    int h  = (idx >> 6) & 15;
    int b  = idx >> 10;

    const __bf16* qbase = qk + (size_t)b * SEQ * 2048 + h * DK;
    const __bf16* kbase = qbase + 1024;
    const __bf16* vbase = vt + (size_t)(b*NHEAD + h) * DK * SEQ;

    // Q as B-fragments for both 16-row halves (lane = q column -> Q row)
    FragBF qb[2][2];
#pragma unroll
    for (int blk = 0; blk < 2; ++blk) {
        const __bf16* qr = qbase + (size_t)(q0 + blk*16 + lr) * 2048 + lh*8;
        qb[blk][0].u[0] = *(const uint4*)qr;
        qb[blk][0].u[1] = *(const uint4*)(qr + 16);
        qb[blk][1].u[0] = *(const uint4*)(qr + 32);
        qb[blk][1].u[1] = *(const uint4*)(qr + 48);
    }

    float mrow[2] = {-1e30f, -1e30f};
    float lrow[2] = {0.f, 0.f};
    v8f o[2][4] = {};                     // per half: O tile 16q x 64d

    int ntiles = (q0 >> 5) + 1;           // kv steps of 32 (same for both halves)
    for (int t = 0; t < ntiles; ++t) {
        int kv0 = t << 5;

        // K as A-fragments: 2 kv subtiles x 2 d-chunks (shared by both halves)
        FragBF ka[2][2];
#pragma unroll
        for (int tt = 0; tt < 2; ++tt) {
            const __bf16* kr = kbase + (size_t)(kv0 + tt*16 + lr) * 2048 + lh*8;
            ka[tt][0].u[0] = *(const uint4*)kr;
            ka[tt][0].u[1] = *(const uint4*)(kr + 16);
            ka[tt][1].u[0] = *(const uint4*)(kr + 32);
            ka[tt][1].u[1] = *(const uint4*)(kr + 48);
        }

        FragBF pa[2];
#pragma unroll
        for (int blk = 0; blk < 2; ++blk) {
            // S^T tiles: ct0 = kv0..+15, ct1 = kv0+16..+31 (rows), q columns
            v8f ct0 = {}, ct1 = {};
            ct0 = wmma_bf16(ka[0][0], qb[blk][0], ct0);
            ct0 = wmma_bf16(ka[0][1], qb[blk][1], ct0);
            ct1 = wmma_bf16(ka[1][0], qb[blk][0], ct1);
            ct1 = wmma_bf16(ka[1][1], qb[blk][1], ct1);

            int q = q0 + blk*16 + lr;
            if (t == ntiles - 1) {        // only last tile crosses diagonal
#pragma unroll
                for (int r = 0; r < 8; ++r) {
                    if (kv0 + r + 8*lh > q)      ct0[r] = -1e30f;
                    if (kv0 + 16 + r + 8*lh > q) ct1[r] = -1e30f;
                }
            }

            // online softmax (per q column; halves combined via xor-16)
            float mx = -1e30f;
#pragma unroll
            for (int r = 0; r < 8; ++r) {
                mx = fmaxf(mx, ct0[r]);
                mx = fmaxf(mx, ct1[r]);
            }
            mx = fmaxf(mx, __shfl_xor(mx, 16, 32));
            float mnew = fmaxf(mrow[blk], mx);

            float p0[8], p1[8], ls = 0.f;
#pragma unroll
            for (int r = 0; r < 8; ++r) {
                p0[r] = __expf((ct0[r] - mnew) * SC);
                p1[r] = __expf((ct1[r] - mnew) * SC);
                ls += p0[r] + p1[r];
            }
            ls += __shfl_xor(ls, 16, 32);
            float alpha = __expf((mrow[blk] - mnew) * SC);
            lrow[blk] = lrow[blk] * alpha + ls;
            mrow[blk] = mnew;

            // P -> A-fragment directly (register-resident, no transpose)
#pragma unroll
            for (int e = 0; e < 8; ++e) {
                pa[blk].v[e]     = (__bf16)p0[e];
                pa[blk].v[e + 8] = (__bf16)p1[e];
            }

            // rescale O rows: alpha for q-row (r+8*lh) lives on lane r+8*lh
#pragma unroll
            for (int r = 0; r < 8; ++r) {
                float ar = __shfl(alpha, r + 8*lh, 32);
#pragma unroll
                for (int n4 = 0; n4 < 4; ++n4) o[blk][n4][r] *= ar;
            }
        }

        // O += P * V : V fragments loaded once, used by both halves
#pragma unroll
        for (int n4 = 0; n4 < 4; ++n4) {
            FragBF vb;
            const __bf16* vr = vbase + (size_t)(n4*16 + lr) * SEQ + kv0 + lh*8;
            vb.u[0] = *(const uint4*)vr;
            vb.u[1] = *(const uint4*)(vr + 16);
            o[0][n4] = wmma_bf16(pa[0], vb, o[0][n4]);
            o[1][n4] = wmma_bf16(pa[1], vb, o[1][n4]);
        }
    }

    // normalize and store (row-major bf16 for the out-proj GEMM)
#pragma unroll
    for (int blk = 0; blk < 2; ++blk) {
#pragma unroll
        for (int r = 0; r < 8; ++r) {
            float lsum = __shfl(lrow[blk], r + 8*lh, 32);
            float inv  = 1.0f / lsum;
            int m = b * SEQ + q0 + blk*16 + r + 8*lh;
#pragma unroll
            for (int n4 = 0; n4 < 4; ++n4)
                attn[(size_t)m * DMODEL + h*DK + n4*16 + lr] =
                    (__bf16)(o[blk][n4][r] * inv);
        }
    }
}

// ---------------------------------------------------------------------------
// Output projection: out[M=4096, N=1024] = attn_bf * Wout^T + b_out  (fp32)
// Block tile 128x256, wave tile 64x64.
// ---------------------------------------------------------------------------
__global__ __launch_bounds__(256) void out_gemm(
    const __bf16* __restrict__ A,    // [4096,1024]
    const __bf16* __restrict__ W,    // [1024,1024]
    const float*  __restrict__ bias, // [1024]
    float* __restrict__ out)         // [4096,1024]
{
    const int K = 1024;
    int lane = threadIdx.x & 31;
    int wv   = threadIdx.x >> 5;
    int lr = lane & 15, lh = lane >> 4;
    int wm = wv >> 2, wn = wv & 3;
    int m0 = blockIdx.y * 128 + wm * 64;
    int n0 = blockIdx.x * 256 + wn * 64;

    const __bf16* arow[4];
    const __bf16* brow[4];
#pragma unroll
    for (int i = 0; i < 4; ++i) {
        arow[i] = A + (size_t)(m0 + i*16 + lr) * K + lh*8;
        brow[i] = W + (size_t)(n0 + i*16 + lr) * K + lh*8;
    }

    v8f acc[4][4] = {};

    for (int k0 = 0; k0 < K; k0 += 32) {
        FragBF a[4], bf[4];
#pragma unroll
        for (int i = 0; i < 4; ++i) {
            a[i].u[0]  = *(const uint4*)(arow[i] + k0);
            a[i].u[1]  = *(const uint4*)(arow[i] + k0 + 16);
            bf[i].u[0] = *(const uint4*)(brow[i] + k0);
            bf[i].u[1] = *(const uint4*)(brow[i] + k0 + 16);
        }
#pragma unroll
        for (int mi = 0; mi < 4; ++mi)
#pragma unroll
            for (int ni = 0; ni < 4; ++ni)
                acc[mi][ni] = wmma_bf16(a[mi], bf[ni], acc[mi][ni]);
    }

#pragma unroll
    for (int mi = 0; mi < 4; ++mi) {
        int mbase = m0 + mi * 16;
#pragma unroll
        for (int ni = 0; ni < 4; ++ni) {
            int n = n0 + ni*16 + lr;
            float bv = bias[n];
#pragma unroll
            for (int r = 0; r < 8; ++r) {
                int m = mbase + r + 8*lh;
                out[(size_t)m * 1024 + n] = acc[mi][ni][r] + bv;
            }
        }
    }
}

// ---------------------------------------------------------------------------
extern "C" void kernel_launch(void* const* d_in, const int* in_sizes, int n_in,
                              void* d_out, int out_size, void* d_ws, size_t ws_size,
                              hipStream_t stream) {
    const float* x     = (const float*)d_in[0];   // [2,2048,1024]
    const float* w_qkv = (const float*)d_in[1];   // [3072,1024]
    const float* b_qkv = (const float*)d_in[2];   // [3072]
    const float* w_out = (const float*)d_in[3];   // [1024,1024]
    const float* b_out = (const float*)d_in[4];   // [1024]
    float* out = (float*)d_out;

    char* ws = (char*)d_ws;
    __bf16* x_bf    = (__bf16*)ws;  ws += (size_t)MROWS * DMODEL * 2;   // 8 MB
    __bf16* wqkv_bf = (__bf16*)ws;  ws += (size_t)NQKV  * DMODEL * 2;   // 6 MB
    __bf16* wout_bf = (__bf16*)ws;  ws += (size_t)DMODEL* DMODEL * 2;   // 2 MB
    __bf16* qk_bf   = (__bf16*)ws;  ws += (size_t)MROWS * 2048   * 2;   // 16 MB
    __bf16* vt_bf   = (__bf16*)ws;  ws += (size_t)BATCH*NHEAD*DK*SEQ*2; // 8 MB
    __bf16* attn_bf = (__bf16*)ws;  ws += (size_t)MROWS * DMODEL * 2;   // 8 MB

    int nx = MROWS * DMODEL;        // 4,194,304
    int nw = NQKV * DMODEL;         // 3,145,728
    int no = DMODEL * DMODEL;       // 1,048,576
    cvt_f32_bf16<<<(nx + 255) / 256, 256, 0, stream>>>(x,     x_bf,    nx);
    cvt_f32_bf16<<<(nw + 255) / 256, 256, 0, stream>>>(w_qkv, wqkv_bf, nw);
    cvt_f32_bf16<<<(no + 255) / 256, 256, 0, stream>>>(w_out, wout_bf, no);

    qkv_gemm<<<dim3(NQKV / 256, MROWS / 128), 256, 0, stream>>>(
        x_bf, wqkv_bf, b_qkv, qk_bf, vt_bf);

    attn_kernel<<<BATCH * NHEAD * (SEQ / 32), 32, 0, stream>>>(
        qk_bf, vt_bf, attn_bf);

    out_gemm<<<dim3(DMODEL / 256, MROWS / 128), 256, 0, stream>>>(
        attn_bf, wout_bf, b_out, out);
}
